// PackDecoder_55748675502544
// MI455X (gfx1250) — compile-verified, matrix operand order checked by reading, not run on hardware
//
#include <hip/hip_runtime.h>
#include <hip/hip_bf16.h>

typedef __attribute__((ext_vector_type(16))) _Float16 v16h;
typedef __attribute__((ext_vector_type(8)))  float    v8f;

#define LEAK 0.01f

__device__ __forceinline__ float leaky(float v) { return v >= 0.f ? v : LEAK * v; }

// ---------------------------------------------------------------------------
// Fragment-packed layout (ISA 7.12.2, 16-bit operands, wave32):
//   element (m|n, k) -> lane = (mn&15) | ((k>>3 & 1)<<4)
//                       e    = (k&7)   | ((k>>4 & 1)<<3)     (e in 0..15 halves)
// A tile is permuted into this layout during LDS staging (free: uint4 stores).
// B (weights) are permuted once in prep_kernel, tiled [k/32][n/16][lane][e].

__device__ __forceinline__ void decode_packed(int i, int Nsub, int& n, int& k) {
    int e    = i & 15;
    int lane = (i >> 4) & 31;
    int rest = i >> 9;
    int ns   = rest % Nsub;
    int kc32 = rest / Nsub;
    n = ns * 16 + (lane & 15);
    k = (e & 7) | ((lane >> 4) << 3) | ((e >> 3) << 4) | (kc32 << 5);
}

// ---------------------------------------------------------------------------
// Weight prep: f32 -> f16, fragment-packed global layouts for WMMA B operands.
__global__ void prep_kernel(const float* __restrict__ wih, const float* __restrict__ whh,
                            const float* __restrict__ c2w, const float* __restrict__ c3w,
                            const float* __restrict__ lasthh,
                            _Float16* __restrict__ wiht, _Float16* __restrict__ whht,
                            _Float16* __restrict__ w2t,  _Float16* __restrict__ w3t,
                            _Float16* __restrict__ h0f) {
    int stride = gridDim.x * blockDim.x;
    int t0 = blockIdx.x * blockDim.x + threadIdx.x;
    // GRU weights: B[k][n] = w[n*512+k], N=1536 (Nsub=96), K=512
    for (int i = t0; i < 512 * 1536; i += stride) {
        int n, k; decode_packed(i, 96, n, k);
        wiht[i] = (_Float16)wih[n * 512 + k];
        whht[i] = (_Float16)whh[n * 512 + k];
    }
    // c3 weights: B[k][o], k = (y*16+x)*128+c, N=256 (Nsub=16), K=32768
    for (int i = t0; i < 32768 * 256; i += stride) {
        int o, k; decode_packed(i, 16, o, k);
        int pix = k >> 7, c = k & 127;
        int y = pix >> 4, x = pix & 15;
        w3t[i] = (_Float16)c3w[((o * 128 + c) * 16 + y) * 16 + x];
    }
    // c2 weights: B[k][co], k = tap*64+ci, N=128 (Nsub=8), K=576
    for (int i = t0; i < 576 * 128; i += stride) {
        int co, k; decode_packed(i, 8, co, k);
        int tap = k >> 6, ci = k & 63;
        int ky = tap / 3, kx = tap % 3;
        w2t[i] = (_Float16)c2w[((co * 64 + ci) * 3 + ky) * 3 + kx];
    }
    // hidden state -> f16 for GRU GEMM
    for (int i = t0; i < 1024 * 512; i += stride)
        h0f[i] = (_Float16)lasthh[i];
}

// ---------------------------------------------------------------------------
// Permuted A-tile staging: source = 64-half contiguous row chunk; two uint4
// stores land each 8-half k-group at its fragment-packed LDS slot.
__device__ __forceinline__ void stage_a_row(uint32_t* sAf_u32, int iRow, int q,
                                            const uint4* src /* may be null */) {
#pragma unroll
    for (int h = 0; h < 2; ++h) {
        int kh8   = q * 16 + h * 8;           // 8-half group base within 64
        int kcl   = kh8 >> 5;                 // which 16x32 k-chunk (0/1)
        int kp    = kh8 & 31;
        int laneD = (iRow & 15) + (((kp >> 3) & 1) << 4);
        int e0    = ((kp >> 4) & 1) << 3;
        uint4 val = src ? src[h] : make_uint4(0u, 0u, 0u, 0u);
        *(uint4*)((_Float16*)sAf_u32 + ((((iRow >> 4) * 2 + kcl) * 32 + laneD) << 4) + e0) = val;
    }
}

// ---------------------------------------------------------------------------
// pre_vec = pre_box @ prebox_w^T + prebox_b  -> x[:, 0:256] (f16)
__global__ void prebox_kernel(const float* __restrict__ pb, const float* __restrict__ pw,
                              const float* __restrict__ pbias, _Float16* __restrict__ x) {
    int b = blockIdx.x;
    int o = threadIdx.x;  // 256
    float v = pb[b * 3 + 0] * pw[o * 3 + 0]
            + pb[b * 3 + 1] * pw[o * 3 + 1]
            + pb[b * 3 + 2] * pw[o * 3 + 2] + pbias[o];
    x[(size_t)b * 512 + o] = (_Float16)v;
}

// ---------------------------------------------------------------------------
// conv1: 2->64 ch, 3x3 pad 1, leaky. Output NHWC f16 (K=18: VALU direct).
__global__ __launch_bounds__(256) void conv1_kernel(const float* __restrict__ hm,
                                                    const float* __restrict__ w1,
                                                    const float* __restrict__ b1,
                                                    _Float16* __restrict__ h1) {
    __shared__ float sW[1152];
    __shared__ float sB[64];
    int b = blockIdx.x >> 5;
    int y = blockIdx.x & 31;
    for (int i = threadIdx.x; i < 1152; i += 256) sW[i] = w1[i];
    if (threadIdx.x < 64) sB[threadIdx.x] = b1[threadIdx.x];
    __syncthreads();
    int x = threadIdx.x & 31;
    int cg = threadIdx.x >> 5;  // 8 output channels per thread
    float p[2][3][3];
#pragma unroll
    for (int ci = 0; ci < 2; ++ci)
#pragma unroll
        for (int dy = 0; dy < 3; ++dy)
#pragma unroll
            for (int dx = 0; dx < 3; ++dx) {
                int iy = y - 1 + dy, ix = x - 1 + dx;
                p[ci][dy][dx] = (iy >= 0 && iy < 32 && ix >= 0 && ix < 32)
                    ? hm[(((size_t)b * 2 + ci) * 32 + iy) * 32 + ix] : 0.f;
            }
    _Float16 outv[8] __attribute__((aligned(16)));
#pragma unroll
    for (int j = 0; j < 8; ++j) {
        int co = cg * 8 + j;
        float acc = sB[co];
        const float* w = &sW[co * 18];
#pragma unroll
        for (int ci = 0; ci < 2; ++ci)
#pragma unroll
            for (int dy = 0; dy < 3; ++dy)
#pragma unroll
                for (int dx = 0; dx < 3; ++dx)
                    acc = fmaf(p[ci][dy][dx], w[(ci * 3 + dy) * 3 + dx], acc);
        outv[j] = (_Float16)leaky(acc);
    }
    *(uint4*)&h1[(((size_t)b * 32 + y) * 32 + x) * 64 + cg * 8] = *(const uint4*)outv;
}

// ---------------------------------------------------------------------------
// conv2 implicit-GEMM: 9 taps x K=64 chunks, WMMA f16->f32.
// Block: 64 output pixels x 128 channels; 8 waves, each 16x64 strip.
__global__ __launch_bounds__(256) void conv2_kernel(const _Float16* __restrict__ h1,
                                                    const _Float16* __restrict__ w2t,
                                                    const float* __restrict__ b2,
                                                    _Float16* __restrict__ h2) {
    __shared__ _Float16 sAf[4096];   // 4 row-tiles x 2 k-chunks x 32 x 16
    __shared__ _Float16 sBf[8192];   // 2 k-chunks x 8 subtiles x 32 x 16
    int bimg = blockIdx.x >> 2;
    int tile = blockIdx.x & 3;
    int tid = threadIdx.x;
    int wave = tid >> 5, lane = tid & 31;
    int r = wave >> 1;      // row tile 0..3
    int chf = wave & 1;     // column half

    int iRow = tid >> 2, q = tid & 3;
    int p  = tile * 64 + iRow;
    int oy = p >> 4, ox = p & 15;

    v8f acc[4] = {};

    for (int t = 0; t < 9; ++t) {
        int ky = t / 3, kx = t % 3;
        // B tile: two contiguous 8KB fragment-packed slabs
        {
            const uint4* s0 = (const uint4*)(w2t + (size_t)(2 * t) * 4096);
            const uint4* s1 = (const uint4*)(w2t + (size_t)(2 * t + 1) * 4096);
            uint4* d = (uint4*)sBf;
#pragma unroll
            for (int u = 0; u < 2; ++u) {
                d[tid * 2 + u]       = s0[tid * 2 + u];
                d[512 + tid * 2 + u] = s1[tid * 2 + u];
            }
        }
        // A tile: 64 patch rows (one NHWC tap = 64 contiguous halves), permuted
        {
            int iy = oy * 2 - 1 + ky, ix = ox * 2 - 1 + kx;
            const uint4* src = nullptr;
            if (iy >= 0 && iy < 32 && ix >= 0 && ix < 32)
                src = (const uint4*)(h1 + (((size_t)bimg * 32 + iy) * 32 + ix) * 64) + q * 2;
            stage_a_row((uint32_t*)sAf, iRow, q, src);
        }
        __syncthreads();
#pragma unroll
        for (int kcl = 0; kcl < 2; ++kcl) {
            v16h a = *(const v16h*)(sAf + (((r * 2 + kcl) * 32 + lane) << 4));
#pragma unroll
            for (int s = 0; s < 4; ++s) {
                int ss = chf * 4 + s;
                v16h b = *(const v16h*)(sBf + (((kcl * 8 + ss) * 32 + lane) << 4));
                acc[s] = __builtin_amdgcn_wmma_f32_16x16x32_f16(false, a, false, b,
                                                                (short)0, acc[s], false, false);
            }
        }
        __syncthreads();
    }
    // bias + leaky, NHWC f16 store
#pragma unroll
    for (int s = 0; s < 4; ++s) {
        int co = (chf * 4 + s) * 16 + (lane & 15);
        float bv = b2[co];
#pragma unroll
        for (int v = 0; v < 8; ++v) {
            int m = r * 16 + ((lane >> 4) << 3) + v;
            int pp = tile * 64 + m;
            float val = leaky(acc[s][v] + bv);
            h2[(((size_t)bimg * 16 + (pp >> 4)) * 16 + (pp & 15)) * 128 + co] = (_Float16)val;
        }
    }
}

// ---------------------------------------------------------------------------
// Generic f16 GEMM: C[M x N] = A[M x K] * Bpacked[K x N] (+bias) (opt leaky).
// Block tile 64x128, K chunks of 64, 8 waves each own a 16x64 strip.
__global__ __launch_bounds__(256) void gemm_f16_kernel(const _Float16* __restrict__ A, int lda,
                                                       const _Float16* __restrict__ Bp, int Nsub,
                                                       int K, const float* __restrict__ bias,
                                                       int do_leaky,
                                                       float* __restrict__ Cf32,
                                                       _Float16* __restrict__ Cf16, int ldc) {
    __shared__ _Float16 sAf[4096];
    __shared__ _Float16 sBf[8192];
    int m0 = blockIdx.y * 64;
    int n_blk = blockIdx.x;
    int tid = threadIdx.x;
    int wave = tid >> 5, lane = tid & 31;
    int r = wave >> 1, chf = wave & 1;

    int iRow = tid >> 2, q = tid & 3;
    const _Float16* arow = A + (size_t)(m0 + iRow) * lda + q * 16;

    v8f acc[4] = {};

    for (int kc = 0; kc < K; kc += 64) {
        int t = kc >> 6;
        const uint4* bs0 = (const uint4*)(Bp + ((size_t)(2 * t) * Nsub + n_blk * 8) * 512);
        const uint4* bs1 = (const uint4*)(Bp + ((size_t)(2 * t + 1) * Nsub + n_blk * 8) * 512);
        // prefetch next chunk's streams into cache
        if (kc + 64 < K) {
            __builtin_prefetch(arow + kc + 64, 0, 1);
            __builtin_prefetch((const _Float16*)bs0 + 2 * (size_t)Nsub * 512 + tid * 16, 0, 1);
        }
        // A chunk: permuted stage (fragment-packed)
        stage_a_row((uint32_t*)sAf, iRow, q, (const uint4*)(arow + kc));
        // B chunk: contiguous copies of pre-packed slabs
        {
            uint4* d = (uint4*)sBf;
#pragma unroll
            for (int u = 0; u < 2; ++u) {
                d[tid * 2 + u]       = bs0[tid * 2 + u];
                d[512 + tid * 2 + u] = bs1[tid * 2 + u];
            }
        }
        __syncthreads();
#pragma unroll
        for (int kcl = 0; kcl < 2; ++kcl) {
            v16h a = *(const v16h*)(sAf + (((r * 2 + kcl) * 32 + lane) << 4));
#pragma unroll
            for (int s = 0; s < 4; ++s) {
                int ss = chf * 4 + s;
                v16h b = *(const v16h*)(sBf + (((kcl * 8 + ss) * 32 + lane) << 4));
                acc[s] = __builtin_amdgcn_wmma_f32_16x16x32_f16(false, a, false, b,
                                                                (short)0, acc[s], false, false);
            }
        }
        __syncthreads();
    }
#pragma unroll
    for (int s = 0; s < 4; ++s) {
        int n = n_blk * 128 + (chf * 4 + s) * 16 + (lane & 15);
        float bv = bias ? bias[n] : 0.f;
#pragma unroll
        for (int v = 0; v < 8; ++v) {
            int m = m0 + r * 16 + ((lane >> 4) << 3) + v;
            float val = acc[s][v] + bv;
            if (do_leaky) val = leaky(val);
            if (Cf32) Cf32[(size_t)m * ldc + n] = val;
            else      Cf16[(size_t)m * ldc + n] = (_Float16)val;
        }
    }
}

// ---------------------------------------------------------------------------
// GRU gate math (f32), writes h_new to both output slots.
__global__ void gates_kernel(const float* __restrict__ gi, const float* __restrict__ gh,
                             const float* __restrict__ lasthh, float* __restrict__ out) {
    int idx = blockIdx.x * blockDim.x + threadIdx.x;
    if (idx >= 1024 * 512) return;
    int b = idx >> 9, j = idx & 511;
    size_t gb = (size_t)b * 1536;
    float i_r = gi[gb + j], i_z = gi[gb + 512 + j], i_n = gi[gb + 1024 + j];
    float h_r = gh[gb + j], h_z = gh[gb + 512 + j], h_n = gh[gb + 1024 + j];
    float rr = 1.f / (1.f + __expf(-(i_r + h_r)));
    float zz = 1.f / (1.f + __expf(-(i_z + h_z)));
    float nn = tanhf(i_n + rr * h_n);
    float h0 = lasthh[idx];
    float hn = (1.f - zz) * nn + zz * h0;
    out[idx] = hn;                 // rnn_out
    out[1024 * 512 + idx] = hn;    // new_hh
}

// ---------------------------------------------------------------------------
extern "C" void kernel_launch(void* const* d_in, const int* in_sizes, int n_in,
                              void* d_out, int out_size, void* d_ws, size_t ws_size,
                              hipStream_t stream) {
    (void)in_sizes; (void)n_in; (void)out_size; (void)ws_size;
    const float* pre_box   = (const float*)d_in[0];
    const float* heightmap = (const float*)d_in[1];
    const float* last_hh   = (const float*)d_in[2];
    const float* prebox_w  = (const float*)d_in[3];
    const float* prebox_b  = (const float*)d_in[4];
    const float* c1_w      = (const float*)d_in[5];
    const float* c1_b      = (const float*)d_in[6];
    const float* c2_w      = (const float*)d_in[7];
    const float* c2_b      = (const float*)d_in[8];
    const float* c3_w      = (const float*)d_in[9];
    const float* c3_b      = (const float*)d_in[10];
    const float* gru_w_ih  = (const float*)d_in[11];
    const float* gru_w_hh  = (const float*)d_in[12];
    const float* gru_b_ih  = (const float*)d_in[13];
    const float* gru_b_hh  = (const float*)d_in[14];
    float* out = (float*)d_out;

    char* ws = (char*)d_ws;
    size_t off = 0;
    auto alloc = [&](size_t bytes) {
        void* pt = ws + off;
        off = (off + bytes + 255) & ~(size_t)255;
        return pt;
    };
    _Float16* x    = (_Float16*)alloc((size_t)1024 * 512 * 2);           // GRU input, f16
    _Float16* h0f  = (_Float16*)alloc((size_t)1024 * 512 * 2);           // hidden, f16
    _Float16* h1   = (_Float16*)alloc((size_t)1024 * 32 * 32 * 64 * 2);  // conv1 out NHWC
    _Float16* h2   = (_Float16*)alloc((size_t)1024 * 16 * 16 * 128 * 2); // conv2 out NHWC
    _Float16* w2t  = (_Float16*)alloc((size_t)576 * 128 * 2);            // fragment-packed
    _Float16* w3t  = (_Float16*)alloc((size_t)32768 * 256 * 2);          // fragment-packed
    _Float16* wiht = (_Float16*)alloc((size_t)512 * 1536 * 2);           // fragment-packed
    _Float16* whht = (_Float16*)alloc((size_t)512 * 1536 * 2);           // fragment-packed
    float*    gi   = (float*)alloc((size_t)1024 * 1536 * 4);
    float*    gh   = (float*)alloc((size_t)1024 * 1536 * 4);

    prep_kernel<<<2048, 256, 0, stream>>>(gru_w_ih, gru_w_hh, c2_w, c3_w, last_hh,
                                          wiht, whht, w2t, w3t, h0f);
    prebox_kernel<<<1024, 256, 0, stream>>>(pre_box, prebox_w, prebox_b, x);
    conv1_kernel<<<32768, 256, 0, stream>>>(heightmap, c1_w, c1_b, h1);
    conv2_kernel<<<4096, 256, 0, stream>>>(h1, w2t, c2_b, h2);
    // conv3 == GEMM (1024 x 32768) * (32768 x 256), +bias, leaky -> x[:, 256:]
    gemm_f16_kernel<<<dim3(2, 16), 256, 0, stream>>>(h2, 32768, w3t, 16, 32768,
                                                     c3_b, 1, nullptr, x + 256, 512);
    // gi = x @ w_ih^T + b_ih   (1024 x 1536)
    gemm_f16_kernel<<<dim3(12, 16), 256, 0, stream>>>(x, 512, wiht, 96, 512,
                                                      gru_b_ih, 0, gi, nullptr, 1536);
    // gh = h0 @ w_hh^T + b_hh  (1024 x 1536)
    gemm_f16_kernel<<<dim3(12, 16), 256, 0, stream>>>(h0f, 512, whht, 96, 512,
                                                      gru_b_hh, 0, gh, nullptr, 1536);
    gates_kernel<<<2048, 256, 0, stream>>>(gi, gh, last_hh, out);
}
